// ARDiscriminator_10058813407269
// MI455X (gfx1250) — matrix operand order repaired and checked by reading, not compiled, not run on hardware
//
#include <hip/hip_runtime.h>

// ---------------------------------------------------------------------------
// AR discriminator for MI455X (gfx1250, wave32, WMMA 16x16x32 f16 -> f32).
//
//   [prep]     fp32 weights -> f16, re-ordered to tap-major K (k' = t*Cin+c)
//   [encoder]  one workgroup per needed frame (only f = 256..511 per batch
//              reach the last-timestep output); 9 conv layers chained in LDS,
//              activations stored position-major with zeroed halos so every
//              WMMA A-fragment is two ds_load_b128.
//   [wavenet]  8 launches on decimated positions t = 511 - m*2^i; H stored
//              [b][m][c] so A-fragments are two global 16B loads; main+gate
//              share A.
//   [finalize] latent[:, :, -1] = feats ; j = judge . feats
// ---------------------------------------------------------------------------

typedef __attribute__((ext_vector_type(16))) _Float16 v16h;
typedef __attribute__((ext_vector_type(8)))  _Float16 v8h;
typedef __attribute__((ext_vector_type(8)))  float    v8f;

#define DEVINL __device__ __forceinline__

constexpr int RS = 136;   // LDS activation row stride (halves): 128 + 8 skew

DEVINL v8f wmma16x16x32(v16h a, v16h b, v8f c) {
  return __builtin_amdgcn_wmma_f32_16x16x32_f16(false, a, false, b,
                                                (short)0, c, false, false);
}

// A fragment: lane half 'hi' selects K offset 16*hi;
// elements e=0..15 -> K = base+0..15 (contiguous, matches B).
// Two contiguous 16B loads.
DEVINL v16h load_a_frag(const _Float16* p) {
  const v8h* q = (const v8h*)p;
  v8h lo = q[0], hi = q[1];
  v16h r;
#pragma unroll
  for (int i = 0; i < 8; ++i) { r[i] = lo[i]; r[i + 8] = hi[i]; }
  return r;
}

// B fragment: 16 contiguous K values for this lane's column: two 16B loads.
DEVINL v16h load_b_frag(const _Float16* __restrict__ p) {
  const v8h* q = (const v8h*)p;
  v8h lo = q[0], hi = q[1];
  v16h r;
#pragma unroll
  for (int i = 0; i < 8; ++i) { r[i] = lo[i]; r[i + 8] = hi[i]; }
  return r;
}

// ---------------------------------------------------------------------------
// Weight prep kernels (one-time, tiny)
// ---------------------------------------------------------------------------
__global__ void cvt_f32_to_f16(const float* __restrict__ src,
                               _Float16* __restrict__ dst, int n) {
  int i = blockIdx.x * 256 + threadIdx.x;
  if (i < n) dst[i] = (_Float16)src[i];
}

// src [CO'][CI][KW] -> dst[co][t*CI + ci]  (tap-major K)
__global__ void prep_conv_w(const float* __restrict__ src,
                            _Float16* __restrict__ dst, int CI, int KW, int n) {
  int i = blockIdx.x * 256 + threadIdx.x;
  if (i >= n) return;
  const int t  = i % KW;
  const int ci = (i / KW) % CI;
  const int co = i / (KW * CI);
  dst[co * (CI * KW) + t * CI + ci] = (_Float16)src[i];
}

// src [8][256][256][2] -> dst[(l*256+c)*512 + tap*256 + cin]
__global__ void prep_wg_w(const float* __restrict__ src,
                          _Float16* __restrict__ dst, int n) {
  int i = blockIdx.x * 256 + threadIdx.x;
  if (i >= n) return;
  const int tap = i & 1;
  const int cin = (i >> 1) & 255;
  const int hc  = i >> 9;          // l*256 + c
  dst[hc * 512 + tap * 256 + cin] = (_Float16)src[i];
}

// ---------------------------------------------------------------------------
// Encoder conv layer as WMMA GEMM (LDS pos-major, halo-padded, stride 2).
//   M = out positions, N = 128, K = 128*KW (tap-major).
//   k-chunk => constant tap t = kk>>7, channel base cb = kk&127,
//   A element = sIn[(2*arow + t)*RS + cb + ...]  (halo absorbs -pad)
// ---------------------------------------------------------------------------
template <int KW>
DEVINL void conv_layer(const _Float16* sIn, _Float16* sOut,
                       const _Float16* __restrict__ W,
                       int out_len, int npad_out, int tid) {
  const int K = 128 * KW;
  const int lane = tid & 31, wave = tid >> 5;
  const int row = lane & 15, hi = lane >> 4;

  // zero the output halo (next layer's padding)
  for (int i = tid; i < npad_out * 128; i += 256) {
    const int r = i >> 7, c = i & 127;
    sOut[r * RS + c] = (_Float16)0.f;
    sOut[(npad_out + out_len + r) * RS + c] = (_Float16)0.f;
  }

  const int mtiles = (out_len + 15) >> 4;
  for (int tile = wave; tile < mtiles * 8; tile += 8) {
    const int mt = tile >> 3, nt = tile & 7;
    const int m0 = mt << 4, n0 = nt << 4;
    int arow = m0 + row;
    if (arow >= out_len) arow = out_len - 1;   // clamped; rows discarded later
    const _Float16* wrow = W + (n0 + row) * K + (hi << 4);
    v8f acc = {};
#pragma unroll 4
    for (int kk = 0; kk < K; kk += 32) {
      const int t  = kk >> 7;
      const int cb = kk & 127;
      const v16h a = load_a_frag(sIn + (2 * arow + t) * RS + cb + (hi << 4));
      const v16h b = load_b_frag(wrow + kk);
      acc = wmma16x16x32(a, b, acc);
    }
#pragma unroll
    for (int v = 0; v < 8; ++v) {
      float r = acc[v];
      r = r > 0.f ? r : 0.2f * r;              // leaky_relu(0.2)
      const int mr = m0 + v + (hi << 3);
      if (mr < out_len) sOut[(mr + npad_out) * RS + n0 + row] = (_Float16)r;
    }
  }
}

// One workgroup = one frame (b, f = 511 - m). 8 waves, 256 threads.
__global__ __launch_bounds__(256)
void encoder_kernel(const float* __restrict__ x,
                    const _Float16* __restrict__ w0,
                    const _Float16* __restrict__ w123,
                    const _Float16* __restrict__ w4567,
                    const _Float16* __restrict__ w8,
                    _Float16* __restrict__ H0) {
  __shared__ _Float16 s_in[256];
  __shared__ _Float16 sA[134 * RS];   // (128 + 2*3 halo) rows
  __shared__ _Float16 sB[70 * RS];    // (64 + 2*3 halo) rows

  const int wg = blockIdx.x;          // 0..2047
  const int b  = wg >> 8;
  const int m  = wg & 255;            // H position index
  const int f  = 511 - m;             // frame within batch
  const int tid  = threadIdx.x;
  const int lane = tid & 31;
  const int wave = tid >> 5;

  s_in[tid] = (_Float16)x[(b * 256 + tid) * 512 + f];
  // conv0 halo (3 rows each side of 128)
  for (int i = tid; i < 3 * 128; i += 256) {
    const int r = i >> 7, c = i & 127;
    sA[r * RS + c] = (_Float16)0.f;
    sA[(131 + r) * RS + c] = (_Float16)0.f;
  }
  __syncthreads();

  // conv0: Cin=1, k=7, s=2, p=3 : 256 -> [128pos][128ch]  (<1% of FLOPs)
  for (int i = tid; i < 128 * 128; i += 256) {
    const int co = i & 127, o = i >> 7;
    float acc = 0.f;
#pragma unroll
    for (int t = 0; t < 7; ++t) {
      const int pos = 2 * o - 3 + t;
      const float v = (pos >= 0 && pos < 256) ? (float)s_in[pos] : 0.f;
      acc += (float)w0[co * 7 + t] * v;
    }
    acc = acc > 0.f ? acc : 0.2f * acc;
    sA[(o + 3) * RS + co] = (_Float16)acc;
  }
  __syncthreads();

  conv_layer<7>(sA, sB, w123 + 0 * 114688, 64, 3, tid); __syncthreads();
  conv_layer<7>(sB, sA, w123 + 1 * 114688, 32, 3, tid); __syncthreads();
  conv_layer<7>(sA, sB, w123 + 2 * 114688, 16, 1, tid); __syncthreads();
  conv_layer<3>(sB, sA, w4567 + 0 * 49152,  8, 1, tid); __syncthreads();
  conv_layer<3>(sA, sB, w4567 + 1 * 49152,  4, 1, tid); __syncthreads();
  conv_layer<3>(sB, sA, w4567 + 2 * 49152,  2, 1, tid); __syncthreads();
  conv_layer<3>(sA, sB, w4567 + 3 * 49152,  1, 0, tid); __syncthreads();

  // L8: 1x1 conv, 128 -> 256, out_len = 1, no activation; write H0[b][m][c]
  {
    const int row = lane & 15, hi = lane >> 4;
    for (int tile = wave; tile < 16; tile += 8) {
      const int n0 = tile << 4;
      const _Float16* wrow = w8 + (n0 + row) * 128 + (hi << 4);
      v8f acc = {};
#pragma unroll
      for (int kk = 0; kk < 128; kk += 32) {
        const v16h a = load_a_frag(sB + kk + (hi << 4));  // row 0 broadcast
        const v16h b = load_b_frag(wrow + kk);
        acc = wmma16x16x32(a, b, acc);
      }
      if (hi == 0)   // D row M=0 = acc[0] of lanes 0..15
        H0[((b << 8) + m) * 256 + n0 + row] = (_Float16)acc[0];
    }
  }
}

// ---------------------------------------------------------------------------
// WaveNet layer i on decimated positions. H[b][m][c] = h_i(511 - m*2^i).
//   GEMM: M = Cz = 2^(7-i), N = 256, K = 512 tap-major:
//   kk<256 -> tap0 (h at 2m'+1), kk>=256 -> tap1 (h at 2m').
// One wave per 16x16 tile; A shared between main and gate WMMAs.
// ---------------------------------------------------------------------------
__global__ __launch_bounds__(256)
void wavenet_layer(const _Float16* __restrict__ Hin,
                   _Float16* __restrict__ Hout,
                   const _Float16* __restrict__ Wm,
                   const _Float16* __restrict__ Wg,
                   float* __restrict__ feats, int layer, int Cz) {
  const int mtiles = (Cz + 15) >> 4;
  const int total  = 8 * 16 * mtiles;
  const int tile = blockIdx.x * 8 + (threadIdx.x >> 5);
  if (tile >= total) return;            // wave-uniform exit
  const int lane = threadIdx.x & 31;
  const int row = lane & 15, hi = lane >> 4;
  int t = tile;
  const int mt = t % mtiles; t /= mtiles;
  const int nt = t & 15;     t >>= 4;
  const int b  = t;
  const int m0 = mt << 4, n0 = nt << 4;

  const _Float16* hb  = Hin + (b << 16);                       // [m][c]
  const _Float16* wmp = Wm + layer * 131072 + (n0 + row) * 512 + (hi << 4);
  const _Float16* wgp = Wg + layer * 131072 + (n0 + row) * 512 + (hi << 4);

  int arow = m0 + row;
  if (arow >= Cz) arow = Cz - 1;        // clamp; padded rows discarded
  v8f accM = {}, accG = {};
#pragma unroll 4
  for (int kk = 0; kk < 512; kk += 32) {
    const int tap = kk >> 8;
    const int cb  = kk & 255;
    const int p   = 2 * arow + 1 - tap;                        // h position idx
    const v16h a  = load_a_frag(hb + (p << 8) + cb + (hi << 4));
    const v16h bm = load_b_frag(wmp + kk);
    const v16h bg = load_b_frag(wgp + kk);
    accM = wmma16x16x32(a, bm, accM);
    accG = wmma16x16x32(a, bg, accG);
  }

  const int c = n0 + row;
#pragma unroll
  for (int v = 0; v < 8; ++v) {
    const int mr = m0 + v + (hi << 3);
    if (mr < Cz) {
      const float z = tanhf(accM[v]) * (1.f / (1.f + __expf(-accG[v])));
      const float hres = (float)Hin[(b << 16) + ((2 * mr) << 8) + c];
      Hout[(b << 16) + (mr << 8) + c] = (_Float16)(hres + z);
      if (mr == 0) {                    // t = 511 contribution
        const int fi = (b << 8) + c;
        feats[fi] = (layer == 0) ? z : feats[fi] + z;
      }
    }
  }
}

// ---------------------------------------------------------------------------
// Finalize: latent[:, :, -1] = feats ; j[b] = sum_c judge[c] * feats[b][c]
// ---------------------------------------------------------------------------
__global__ __launch_bounds__(256)
void finalize_kernel(const float* __restrict__ feats,
                     const float* __restrict__ judge_w,
                     float* __restrict__ out) {
  __shared__ float red[256];
  const int b = blockIdx.x, c = threadIdx.x;
  const float v = feats[b * 256 + c];
  out[b * 256 + c] = v;                           // latent flat [B, C, 1]
  red[c] = v * judge_w[c];
  __syncthreads();
  for (int s = 128; s > 0; s >>= 1) {
    if (c < s) red[c] += red[c + s];
    __syncthreads();
  }
  if (c == 0) out[2048 + b] = red[0];             // j flat [B, 1, 1]
}

// ---------------------------------------------------------------------------
// Host-side orchestration
// ---------------------------------------------------------------------------
extern "C" void kernel_launch(void* const* d_in, const int* in_sizes, int n_in,
                              void* d_out, int out_size, void* d_ws,
                              size_t ws_size, hipStream_t stream) {
  (void)in_sizes; (void)n_in; (void)out_size; (void)ws_size;
  const float* x     = (const float*)d_in[0];
  const float* w0f   = (const float*)d_in[1];
  const float* w123f = (const float*)d_in[2];
  const float* w45f  = (const float*)d_in[3];
  const float* w8f   = (const float*)d_in[4];
  const float* wmf   = (const float*)d_in[5];
  const float* wgf   = (const float*)d_in[6];
  const float* wjf   = (const float*)d_in[7];

  // workspace layout (bytes), 32B aligned; total ~7.1 MB
  constexpr size_t OFF_W0    = 0;                       //   896 f16
  constexpr size_t OFF_W123  = 2048;                    // 344064 f16
  constexpr size_t OFF_W4567 = OFF_W123  + 688128;      // 196608 f16
  constexpr size_t OFF_W8    = OFF_W4567 + 393216;      //  32768 f16
  constexpr size_t OFF_WM    = OFF_W8    + 65536;       // 1048576 f16
  constexpr size_t OFF_WG    = OFF_WM    + 2097152;     // 1048576 f16
  constexpr size_t OFF_H0    = OFF_WG    + 2097152;     // 8*256*256 f16
  constexpr size_t OFF_H1    = OFF_H0    + 1048576;
  constexpr size_t OFF_FEATS = OFF_H1    + 1048576;     // 8*256 f32

  char* ws = (char*)d_ws;
  _Float16* w0h   = (_Float16*)(ws + OFF_W0);
  _Float16* w123h = (_Float16*)(ws + OFF_W123);
  _Float16* w45h  = (_Float16*)(ws + OFF_W4567);
  _Float16* w8h   = (_Float16*)(ws + OFF_W8);
  _Float16* wmh   = (_Float16*)(ws + OFF_WM);
  _Float16* wgh   = (_Float16*)(ws + OFF_WG);
  _Float16* Hbuf[2] = {(_Float16*)(ws + OFF_H0), (_Float16*)(ws + OFF_H1)};
  float* feats = (float*)(ws + OFF_FEATS);

  cvt_f32_to_f16<<<(896 + 255) / 256, 256, 0, stream>>>(w0f, w0h, 896);
  cvt_f32_to_f16<<<(32768 + 255) / 256, 256, 0, stream>>>(w8f, w8h, 32768);
  prep_conv_w<<<(344064 + 255) / 256, 256, 0, stream>>>(w123f, w123h, 128, 7,
                                                        344064);
  prep_conv_w<<<(196608 + 255) / 256, 256, 0, stream>>>(w45f, w45h, 128, 3,
                                                        196608);
  prep_wg_w<<<(1048576 + 255) / 256, 256, 0, stream>>>(wmf, wmh, 1048576);
  prep_wg_w<<<(1048576 + 255) / 256, 256, 0, stream>>>(wgf, wgh, 1048576);

  // Encoder: only frames 511-m, m=0..255, per batch feed the final output.
  encoder_kernel<<<2048, 256, 0, stream>>>(x, w0h, w123h, w45h, w8h, Hbuf[0]);

  // Decimated WaveNet stack.
  for (int i = 0; i < 8; ++i) {
    const int Cz = 128 >> i;
    const int total = 8 * 16 * ((Cz + 15) / 16);
    wavenet_layer<<<(total + 7) / 8, 256, 0, stream>>>(
        Hbuf[i & 1], Hbuf[(i + 1) & 1], wmh, wgh, feats, i, Cz);
  }

  finalize_kernel<<<8, 256, 0, stream>>>(feats, wjf, (float*)d_out);
}
